// DynamicGRU_34402688041275
// MI455X (gfx1250) — matrix operand order repaired and checked by reading, not compiled, not run on hardware
//
#include <hip/hip_runtime.h>

typedef __attribute__((ext_vector_type(16))) __bf16 v16bf;
typedef __attribute__((ext_vector_type(8)))  __bf16 v8bf;
typedef __attribute__((ext_vector_type(8)))  float  v8f;

#define Bsz 256
#define Ssz 512
#define Isz 512
#define Hsz 512
#define K2  1024   // fused K: [h | x_t]
#define APAD 1032  // padded LDS row stride (1024 + 8) -> conflict-free 16-row b128 reads
#define SLDS_BYTES (32 * APAD * 2)  // 66048 B dynamic LDS per block

// ---------------- prologue kernels ----------------

// fp32 x -> bf16 x, 4 elements per thread
__global__ __launch_bounds__(256) void k_cvt_x(const float* __restrict__ x,
                                               __bf16* __restrict__ xb, int n4) {
  int i = blockIdx.x * blockDim.x + threadIdx.x;
  if (i >= n4) return;
  const float4 v = reinterpret_cast<const float4*>(x)[i];
  __bf16* o = xb + (size_t)i * 4;
  o[0] = (__bf16)v.x; o[1] = (__bf16)v.y; o[2] = (__bf16)v.z; o[3] = (__bf16)v.w;
}

// build combined bf16 weight matrix Wc[3H][1024] = [W_hh | W_ih] row-wise
__global__ __launch_bounds__(256) void k_prep_w(const float* __restrict__ wih,
                                                const float* __restrict__ whh,
                                                __bf16* __restrict__ wc) {
  int i = blockIdx.x * blockDim.x + threadIdx.x;   // over 3H*1024
  if (i >= 3 * Hsz * K2) return;
  int row = i >> 10, k = i & (K2 - 1);
  float v = (k < Hsz) ? whh[(size_t)row * Hsz + k]
                      : wih[(size_t)row * Isz + (k - Hsz)];
  wc[i] = (__bf16)v;
}

// zero-init h state (fp32 + bf16 mirror), buffer 0 only (read at t=0)
__global__ __launch_bounds__(256) void k_init_h(float* __restrict__ hf,
                                                __bf16* __restrict__ hb) {
  int i = blockIdx.x * blockDim.x + threadIdx.x;
  if (i >= Bsz * Hsz) return;
  hf[i] = 0.0f;
  hb[i] = (__bf16)0.0f;
}

__global__ __launch_bounds__(256) void k_copy_hlast(const float* __restrict__ hf,
                                                    float* __restrict__ dst) {
  int i = blockIdx.x * blockDim.x + threadIdx.x;
  if (i < Bsz * Hsz) dst[i] = hf[i];
}

// ---------------- recurrent step kernel ----------------
// grid = (Hsz/32, Bsz/32), block = 128 threads = 4 wave32
// A panel ([h_t | x_t] rows for this block, 32 x 1024 bf16) is staged in LDS
// via async global->LDS copies; WMMA A-fragments then come from ds_load_b128.
// B (weights) streams from global (3 MB combined matrix stays L2-resident).
__global__ __launch_bounds__(128) void k_step(
    int t,
    const __bf16* __restrict__ xb,     // [B,S,I] bf16
    const __bf16* __restrict__ wc,     // [3H,1024] bf16
    const float*  __restrict__ bih,    // [3H]
    const float*  __restrict__ bhh,    // [3H]
    const __bf16* __restrict__ hb_in,  // [B,H] bf16
    const float*  __restrict__ hf_in,  // [B,H] f32
    __bf16* __restrict__ hb_out,
    float*  __restrict__ hf_out,
    float*  __restrict__ out)          // [B,S,H] f32
{
  extern __shared__ __bf16 sA[];       // [32][APAD]

  const int tid  = threadIdx.x;
  const int lane = tid & 31;
  const int w    = tid >> 5;
  const int l15  = lane & 15;
  const int kh   = lane >> 4;                      // lane half (0/1)
  const int m0   = blockIdx.y * 32;                // block batch-row base
  const int m_base = m0 + (w >> 1) * 16;           // wave batch rows
  const int n_base = blockIdx.x * 32 + (w & 1) * 16; // wave hidden cols
  const int bn = n_base + l15;                     // weight row this lane loads

  // ---- stage A panel into LDS with async copies (ASYNCcnt path) ----
  // thread tid owns the 16B chunk at element offset tid*8 of every row:
  //   tid < 64  -> k in [0,512)   from h_t (bf16 mirror)
  //   tid >= 64 -> k in [512,1024) from x[:, t, :]
  {
    const unsigned ldsbase = (unsigned)(size_t)sA;   // LDS aperture: bits[31:0] = LDS offset
    const int kc = tid * 8;
    const bool isH = (kc < Hsz);
    #pragma unroll
    for (int row = 0; row < 32; ++row) {
      const __bf16* g = isH
          ? (hb_in + (size_t)(m0 + row) * Hsz + kc)
          : (xb + ((size_t)(m0 + row) * Ssz + t) * Isz + (kc - Hsz));
      const unsigned loff = ldsbase + (unsigned)((row * APAD + kc) * 2);
      asm volatile("global_load_async_to_lds_b128 %0, %1, off"
                   :: "v"(loff), "v"(g) : "memory");
    }
  }

  // warm L2 for next timestep's x rows while this step computes
  if (t + 1 < Ssz) {
    __builtin_prefetch(
        xb + ((size_t)(m0 + (tid & 31)) * Ssz + (t + 1)) * Isz + ((tid >> 5) * 128),
        0, 1);
  }

#if __has_builtin(__builtin_amdgcn_s_wait_asynccnt)
  __builtin_amdgcn_s_wait_asynccnt(0);
#else
  asm volatile("s_wait_asynccnt 0x0" ::: "memory");
#endif
  __syncthreads();

  // ---- WMMA main loops ----
  v8f acc_r = {}, acc_z = {}, acc_nh = {}, acc_nx = {};

  const __bf16* __restrict__ sArow  = sA + (size_t)((w >> 1) * 16 + l15) * APAD;
  const __bf16* __restrict__ brow_r = wc + (size_t)(0 * Hsz + bn) * K2;
  const __bf16* __restrict__ brow_z = wc + (size_t)(1 * Hsz + bn) * K2;
  const __bf16* __restrict__ brow_n = wc + (size_t)(2 * Hsz + bn) * K2;

  union Afrag { v16bf v; v8bf h[2]; };

  // recurrent half: K = 0..511 (n-gate -> acc_nh)
  #pragma unroll 4
  for (int k0 = 0; k0 < Hsz; k0 += 32) {
    Afrag a;
    a.h[0] = *(const v8bf*)(sArow + k0 + kh * 16);
    a.h[1] = *(const v8bf*)(sArow + k0 + 8 + kh * 16);
    v16bf br  = *(const v16bf*)(brow_r + k0 + kh * 16);
    v16bf bz  = *(const v16bf*)(brow_z + k0 + kh * 16);
    v16bf bn_ = *(const v16bf*)(brow_n + k0 + kh * 16);
    acc_r  = __builtin_amdgcn_wmma_f32_16x16x32_bf16(false, a.v, false, br,  (short)0, acc_r,  false, false);
    acc_z  = __builtin_amdgcn_wmma_f32_16x16x32_bf16(false, a.v, false, bz,  (short)0, acc_z,  false, false);
    acc_nh = __builtin_amdgcn_wmma_f32_16x16x32_bf16(false, a.v, false, bn_, (short)0, acc_nh, false, false);
  }

  // input half: K = 512..1023 (n-gate -> acc_nx)
  #pragma unroll 4
  for (int k0 = 0; k0 < Isz; k0 += 32) {
    Afrag a;
    a.h[0] = *(const v8bf*)(sArow + Hsz + k0 + kh * 16);
    a.h[1] = *(const v8bf*)(sArow + Hsz + k0 + 8 + kh * 16);
    v16bf br  = *(const v16bf*)(brow_r + Hsz + k0 + kh * 16);
    v16bf bz  = *(const v16bf*)(brow_z + Hsz + k0 + kh * 16);
    v16bf bn_ = *(const v16bf*)(brow_n + Hsz + k0 + kh * 16);
    acc_r  = __builtin_amdgcn_wmma_f32_16x16x32_bf16(false, a.v, false, br,  (short)0, acc_r,  false, false);
    acc_z  = __builtin_amdgcn_wmma_f32_16x16x32_bf16(false, a.v, false, bz,  (short)0, acc_z,  false, false);
    acc_nx = __builtin_amdgcn_wmma_f32_16x16x32_bf16(false, a.v, false, bn_, (short)0, acc_nx, false, false);
  }

  // ---- gate math (fp32) ----
  // lane's C/D elements: M = r + 8*kh, N = l15 (per 16x16 f32 C/D layout)
  const float biasr  = bih[bn] + bhh[bn];
  const float biasz  = bih[Hsz + bn] + bhh[Hsz + bn];
  const float biasnx = bih[2 * Hsz + bn];
  const float biasnh = bhh[2 * Hsz + bn];
  const int n = n_base + l15;

  #pragma unroll
  for (int r = 0; r < 8; ++r) {
    const int m = m_base + r + kh * 8;
    const float rg = 1.0f / (1.0f + __expf(-(acc_r[r] + biasr)));
    const float zg = 1.0f / (1.0f + __expf(-(acc_z[r] + biasz)));
    const float ng = tanhf(acc_nx[r] + biasnx + rg * (acc_nh[r] + biasnh));
    const size_t hidx = (size_t)m * Hsz + n;
    const float hold = hf_in[hidx];
    const float hnew = (1.0f - zg) * hold + zg * ng;
    out[((size_t)m * Ssz + t) * Hsz + n] = hnew;
    hf_out[hidx] = hnew;
    hb_out[hidx] = (__bf16)hnew;
  }
}

// ---------------- host launcher ----------------
extern "C" void kernel_launch(void* const* d_in, const int* in_sizes, int n_in,
                              void* d_out, int out_size, void* d_ws, size_t ws_size,
                              hipStream_t stream) {
  const float* x   = (const float*)d_in[0];
  const float* wih = (const float*)d_in[1];
  const float* whh = (const float*)d_in[2];
  const float* bih = (const float*)d_in[3];
  const float* bhh = (const float*)d_in[4];
  float* out = (float*)d_out;

  // workspace layout
  char* ws = (char*)d_ws;
  size_t off = 0;
  __bf16* xb  = (__bf16*)(ws + off); off += (size_t)Bsz * Ssz * Isz * 2;  // 128 MB
  __bf16* wc  = (__bf16*)(ws + off); off += (size_t)3 * Hsz * K2 * 2;     // 3 MB
  float*  hf0 = (float*)(ws + off);  off += (size_t)Bsz * Hsz * 4;
  float*  hf1 = (float*)(ws + off);  off += (size_t)Bsz * Hsz * 4;
  __bf16* hb0 = (__bf16*)(ws + off); off += (size_t)Bsz * Hsz * 2;
  __bf16* hb1 = (__bf16*)(ws + off); off += (size_t)Bsz * Hsz * 2;

  const int nx4 = (Bsz * Ssz * Isz) / 4;
  k_cvt_x<<<nx4 / 256, 256, 0, stream>>>(x, xb, nx4);
  k_prep_w<<<(3 * Hsz * K2) / 256, 256, 0, stream>>>(wih, whh, wc);
  k_init_h<<<(Bsz * Hsz) / 256, 256, 0, stream>>>(hf0, hb0);

  // sequential scan: ping-pong h buffers (step t reads buf[t&1], writes buf[(t+1)&1])
  for (int t = 0; t < Ssz; ++t) {
    const __bf16* hbi = (t & 1) ? hb1 : hb0;
    const float*  hfi = (t & 1) ? hf1 : hf0;
    __bf16* hbo = (t & 1) ? hb0 : hb1;
    float*  hfo = (t & 1) ? hf0 : hf1;
    k_step<<<dim3(Hsz / 32, Bsz / 32), 128, SLDS_BYTES, stream>>>(
        t, xb, wc, bih, bhh, hbi, hfi, hbo, hfo, out);
  }

  // t=511 (odd) wrote hf0 -> h_last
  k_copy_hlast<<<(Bsz * Hsz) / 256, 256, 0, stream>>>(hf0, out + (size_t)Bsz * Ssz * Hsz);
}
